// SelfAttentivePooling_4329327034748
// MI455X (gfx1250) — compile-verified, matrix operand order checked by reading, not run on hardware
//
#include <hip/hip_runtime.h>
#include <hip/hip_bf16.h>
#include <math.h>

// Problem constants (from the reference): B=32, T=2048, D=512.
#define BB   32
#define TT   2048
#define DD   512
#define ROWS 128           // rows (b,t pairs) per workgroup in the scores kernel
#define NRT  8             // row tiles (16 rows each) per workgroup
#define CTW  4             // col tiles per wave (8 waves x 4 = 32 col tiles = D)

typedef __attribute__((ext_vector_type(16))) __bf16        v16bf;
typedef __attribute__((ext_vector_type(8)))  float         v8f;
typedef __attribute__((ext_vector_type(8)))  unsigned int  v8u;

#if __has_builtin(__builtin_amdgcn_tanhf)
  #define TANHF(x) __builtin_amdgcn_tanhf(x)     // hardware V_TANH_F32
#else
  #define TANHF(x) tanhf(x)
#endif

// round-to-nearest-even fp32 -> bf16 (bit pattern as u16)
static __device__ __forceinline__ unsigned short f32_to_bf16(float f) {
    unsigned int u = __builtin_bit_cast(unsigned int, f);
    u += 0x7FFFu + ((u >> 16) & 1u);
    return (unsigned short)(u >> 16);
}

// ---------------------------------------------------------------------------
// Kernel 0: W (fp32, D x D, row = output dim e, col = input dim d) -> bf16
// ---------------------------------------------------------------------------
__global__ __launch_bounds__(256) void sap_convert_w(
    const float* __restrict__ W, unsigned short* __restrict__ Wbf, int n)
{
    int i = blockIdx.x * 256 + threadIdx.x;
    if (i < n) Wbf[i] = f32_to_bf16(W[i]);
}

// ---------------------------------------------------------------------------
// Kernel 1: scores[b*T + t] = attention . tanh(W @ xs[b,t] + bias)
// GEMM M=B*T=65536, N=512, K=512 via v_wmma_f32_16x16x32_bf16.
// Register-blocked: each wave holds 8 row-tile accumulators so every B (W)
// fragment fetched from L2 feeds 8 WMMAs -> B traffic 2GB -> 256MB.
// ---------------------------------------------------------------------------
__global__ __launch_bounds__(256) void sap_scores(
    const float*          __restrict__ xs,     // [B*T, D] fp32
    const unsigned short* __restrict__ Wbf,    // [D, D] bf16 bits (row e, col d)
    const float*          __restrict__ bias,   // [D]
    const float*          __restrict__ attn,   // [D]
    float*                __restrict__ scores) // [B*T]
{
    __shared__ unsigned int ldsA[ROWS * DD / 2];  // 128 KB: bf16 pairs [row][k]
    __shared__ float        ldsS[ROWS];

    const int tid   = threadIdx.x;
    const long row0 = (long)blockIdx.x * ROWS;

    if (tid < ROWS) ldsS[tid] = 0.0f;

    // Stage xs tile (128 x 512 fp32) -> bf16 in LDS. 32768 pairs / 256 thr.
    {
        const float2* xb = (const float2*)(xs + row0 * DD);
        #pragma unroll 4
        for (int i = 0; i < (ROWS * DD / 2) / 256; ++i) {
            int p = i * 256 + tid;                    // bf16-pair index
            float2 v = xb[p];
            ldsA[p] = (unsigned int)f32_to_bf16(v.x)
                    | ((unsigned int)f32_to_bf16(v.y) << 16);
        }
    }
    __syncthreads();

    const int wave = tid >> 5;
    const int lane = tid & 31;
    const int g    = lane >> 4;      // lane half-group
    const int l15  = lane & 15;      // A: row-in-tile; B/C/D: col-in-tile

    // Per-lane A base (uint = bf16 pair); row stride = 256 uints.
    const unsigned int* abase = ldsA + l15 * (DD / 2);

    for (int jt = 0; jt < CTW; ++jt) {                // this wave's col tiles
        const int ncol = (wave * CTW + jt) * 16 + l15;
        const unsigned short* brow = Wbf + (long)ncol * DD;

        v8f acc[NRT];
        #pragma unroll
        for (int rt = 0; rt < NRT; ++rt) acc[rt] = (v8f){};

        for (int kb = 0; kb < DD; kb += 32) {         // K loop
            // B fragment once per k-step (L2), reused by 8 row tiles.
            uint4 b0 = *(const uint4*)(brow + kb + g * 8);
            uint4 b1 = *(const uint4*)(brow + kb + 16 + g * 8);
            v8u bu;
            bu[0]=b0.x; bu[1]=b0.y; bu[2]=b0.z; bu[3]=b0.w;
            bu[4]=b1.x; bu[5]=b1.y; bu[6]=b1.z; bu[7]=b1.w;
            v16bf bv = __builtin_bit_cast(v16bf, bu);

            #pragma unroll
            for (int rt = 0; rt < NRT; ++rt) {
                const unsigned int* arow = abase + rt * 16 * (DD / 2);
                uint4 a0 = *(const uint4*)(arow + (kb + g * 8) / 2);
                uint4 a1 = *(const uint4*)(arow + (kb + 16 + g * 8) / 2);
                v8u au;
                au[0]=a0.x; au[1]=a0.y; au[2]=a0.z; au[3]=a0.w;
                au[4]=a1.x; au[5]=a1.y; au[6]=a1.z; au[7]=a1.w;
                v16bf av = __builtin_bit_cast(v16bf, au);
                acc[rt] = __builtin_amdgcn_wmma_f32_16x16x32_bf16(
                              false, av, false, bv, (short)0, acc[rt], false, false);
            }
        }

        // Epilogue: tanh(h + bias) * attention, reduce over the 16 lanes
        // (distinct N per lane), atomically fold into per-row scores.
        // C/D layout: lane l15 = N; vgpr r = row r (lanes 0-15) / r+8 (16-31).
        const float bn = bias[ncol];
        const float an = attn[ncol];
        #pragma unroll
        for (int rt = 0; rt < NRT; ++rt) {
            #pragma unroll
            for (int r = 0; r < 8; ++r) {
                float v = TANHF(acc[rt][r] + bn) * an;
                v += __shfl_xor(v, 1, 32);
                v += __shfl_xor(v, 2, 32);
                v += __shfl_xor(v, 4, 32);
                v += __shfl_xor(v, 8, 32);
                if (l15 == 0)
                    atomicAdd(&ldsS[rt * 16 + g * 8 + r], v);   // ds_add_f32
            }
        }
    }
    __syncthreads();

    if (tid < ROWS) scores[row0 + tid] = ldsS[tid];
}

// ---------------------------------------------------------------------------
// Kernel 2: per-batch masked softmax over T + alpha-weighted sum of xs.
// xs (128 MiB) should still be L2-resident from kernel 1 (192 MB L2).
// ---------------------------------------------------------------------------
__global__ __launch_bounds__(256) void sap_pool(
    const float* __restrict__ xs,      // [B, T, D]
    const int*   __restrict__ xs_len,  // [B]
    const float* __restrict__ scores,  // [B, T]
    float*       __restrict__ out)     // [B, D]
{
    __shared__ float salpha[TT];   // 8 KB
    __shared__ float wmax[8];
    __shared__ float wsum[8];

    const int b    = blockIdx.x;
    const int tid  = threadIdx.x;
    const int lane = tid & 31;
    const int wv   = tid >> 5;
    const int len  = xs_len[b];
    const float* srow = scores + (long)b * TT;

    // 1) block max over valid frames
    float m = -INFINITY;
    for (int t = tid; t < len; t += 256) m = fmaxf(m, srow[t]);
    #pragma unroll
    for (int off = 16; off >= 1; off >>= 1) m = fmaxf(m, __shfl_xor(m, off, 32));
    if (lane == 0) wmax[wv] = m;
    __syncthreads();
    float bm = wmax[0];
    #pragma unroll
    for (int w = 1; w < 8; ++w) bm = fmaxf(bm, wmax[w]);

    // 2) exp + block sum (masked frames get 0)
    float s = 0.0f;
    for (int t = tid; t < TT; t += 256) {
        float e = (t < len) ? __expf(srow[t] - bm) : 0.0f;
        salpha[t] = e;
        s += e;
    }
    #pragma unroll
    for (int off = 16; off >= 1; off >>= 1) s += __shfl_xor(s, off, 32);
    if (lane == 0) wsum[wv] = s;
    __syncthreads();
    float tot = 0.0f;
    #pragma unroll
    for (int w = 0; w < 8; ++w) tot += wsum[w];
    const float inv = 1.0f / tot;

    // 3) out[b, d] = (1/sum) * sum_t alpha[t] * xs[b,t,d]; thread owns d=2*tid,+1
    float2 acc = make_float2(0.0f, 0.0f);
    const float2* xrow = (const float2*)(xs + (long)b * TT * DD);
    for (int t = 0; t < len; ++t) {
        float a = salpha[t];
        float2 v = xrow[(long)t * (DD / 2) + tid];
        acc.x += a * v.x;
        acc.y += a * v.y;
    }
    ((float2*)out)[b * (DD / 2) + tid] = make_float2(acc.x * inv, acc.y * inv);
}

// ---------------------------------------------------------------------------
// Launcher. Inputs: xs[B*T*D] f32, xs_len[B] i32, W[D*D] f32, b[D] f32,
// attention[D] f32. Output: [B*D] f32.
// ws layout: [0, 512KB) W_bf16 ; [512KB, 512KB+256KB) scores fp32.
// ---------------------------------------------------------------------------
extern "C" void kernel_launch(void* const* d_in, const int* in_sizes, int n_in,
                              void* d_out, int out_size, void* d_ws, size_t ws_size,
                              hipStream_t stream) {
    const float* xs     = (const float*)d_in[0];
    const int*   xs_len = (const int*)  d_in[1];
    const float* W      = (const float*)d_in[2];
    const float* bias   = (const float*)d_in[3];
    const float* attn   = (const float*)d_in[4];
    float*       out    = (float*)d_out;

    unsigned short* Wbf    = (unsigned short*)d_ws;                       // 512 KB
    float*          scores = (float*)((char*)d_ws + (size_t)DD * DD * 2); // 256 KB

    // 0) W -> bf16 (262144 elements)
    sap_convert_w<<<(DD * DD + 255) / 256, 256, 0, stream>>>(W, Wbf, DD * DD);

    // 1) fused GEMM + tanh + attention dot -> scores (512 workgroups)
    sap_scores<<<(BB * TT) / ROWS, 256, 0, stream>>>(xs, Wbf, bias, attn, scores);

    // 2) masked softmax + weighted pooling
    sap_pool<<<BB, 256, 0, stream>>>(xs, xs_len, scores, out);
}
